// ClusteringLayer_16587163697959
// MI455X (gfx1250) — matrix operand order repaired and checked by reading, not compiled, not run on hardware
//
#include <hip/hip_runtime.h>

typedef __attribute__((ext_vector_type(16))) __bf16 v16bf;
typedef __attribute__((ext_vector_type(8)))  float  v8f;

static constexpr int NB = 16384;  // batch rows
static constexpr int ND = 512;    // feature dim
static constexpr int NK = 1024;   // clusters

// Pack two f32 into two bf16 (round-to-nearest, ties away) in 3 VALU:
// two v_add_u32 + one v_perm_b32 selecting the high halves.
__device__ __forceinline__ unsigned pack2bf(float a, float b) {
    unsigned ua = __float_as_uint(a) + 0x8000u;
    unsigned ub = __float_as_uint(b) + 0x8000u;
    // result = { bf16(b) , bf16(a) }  (b in high 16, a in low 16)
    return __builtin_amdgcn_perm(ub, ua, 0x07060302u);
}

union AFrag { uint4 u4[2]; v16bf v; };

// ---------------- row squared-norms: one wave per row ----------------
__global__ void rownorm_kernel(const float* __restrict__ src,
                               float* __restrict__ dst, int rows) {
    int wave = (int)((blockIdx.x * blockDim.x + threadIdx.x) >> 5);
    int lane = threadIdx.x & 31;
    if (wave >= rows) return;
    const float4* p = (const float4*)(src + (size_t)wave * ND);
    float s = 0.f;
#pragma unroll
    for (int j = 0; j < ND / 4 / 32; ++j) {
        float4 v = p[lane + 32 * j];
        s += v.x * v.x + v.y * v.y + v.z * v.z + v.w * v.w;
    }
#pragma unroll
    for (int off = 16; off > 0; off >>= 1) s += __shfl_down(s, off, 32);
    if (lane == 0) dst[wave] = s;
}

__global__ void zero_kernel(float* p) { *p = 0.f; }

// ---------------- fused GEMM + t-student numerator + global-sum ----------------
// block = 256 threads (8 waves); block tile = 128(M) x 128(N); K-step = 32 (bf16 WMMA)
// Double-buffered LDS staging, register prefetch of the next K-slice.
__global__ void __launch_bounds__(256)
tstudent_gemm_kernel(const float* __restrict__ x,
                     const float* __restrict__ c,
                     const float* __restrict__ x2,
                     const float* __restrict__ c2,
                     float* __restrict__ out,
                     float* __restrict__ gsum) {
    __shared__ __align__(16) unsigned short ldsA[2][128 * 32];  // 2 x 8 KB
    __shared__ __align__(16) unsigned short ldsB[2][128 * 32];  // 2 x 8 KB

    const int tid  = threadIdx.x;
    const int lane = tid & 31;
    const int wave = tid >> 5;
    const int wm   = wave & 3;   // 0..3 : 32-row slice
    const int wn   = wave >> 2;  // 0..1 : 64-col slice
    const int half = lane >> 4;
    const int l16  = lane & 15;

    const int m0 = blockIdx.y * 128;
    const int n0 = blockIdx.x * 128;

    // cooperative tile loader: 8 threads per row, float4 each; rows lr+{0,32,64,96}
    const int lr = tid >> 3;        // 0..31
    const int lc = (tid & 7) * 4;   // f32 column 0,4,...,28
    const float* xrow = x + (size_t)(m0 + lr) * ND + lc;
    const float* crow = c + (size_t)(n0 + lr) * ND + lc;

    v8f acc[2][4];
#pragma unroll
    for (int a = 0; a < 2; ++a)
#pragma unroll
        for (int b = 0; b < 4; ++b) acc[a][b] = v8f{};

    // prologue: fetch K-slice 0 into registers
    float4 ra[4], rb[4];
#pragma unroll
    for (int i = 0; i < 4; ++i) {
        ra[i] = *(const float4*)(xrow + (size_t)i * 32 * ND);
        rb[i] = *(const float4*)(crow + (size_t)i * 32 * ND);
    }

    int buf = 0;
    for (int kk = 0; kk < ND; kk += 32) {
        // convert + stage current slice into lds[buf]
#pragma unroll
        for (int i = 0; i < 4; ++i) {
            int row = i * 32 + lr;
            uint2 pa, pb;
            pa.x = pack2bf(ra[i].x, ra[i].y);
            pa.y = pack2bf(ra[i].z, ra[i].w);
            pb.x = pack2bf(rb[i].x, rb[i].y);
            pb.y = pack2bf(rb[i].z, rb[i].w);
            *(uint2*)&ldsA[buf][row * 32 + lc] = pa;
            *(uint2*)&ldsB[buf][row * 32 + lc] = pb;
        }
        // prefetch next K-slice into registers (in flight during WMMA below)
        if (kk + 32 < ND) {
#pragma unroll
            for (int i = 0; i < 4; ++i) {
                ra[i] = *(const float4*)(xrow + (size_t)i * 32 * ND + kk + 32);
                rb[i] = *(const float4*)(crow + (size_t)i * 32 * ND + kk + 32);
            }
        }
        __syncthreads();

        // A fragments: ISA 16-bit A 16x32 layout — lane (m=l16, khalf=half) holds
        // K in [8*half, 8*half+8) and [16+8*half, 16+8*half+8)
        AFrag afr[2];
#pragma unroll
        for (int a = 0; a < 2; ++a) {
            int row = wm * 32 + a * 16 + l16;
            afr[a].u4[0] = *(const uint4*)&ldsA[buf][row * 32 + 8 * half];
            afr[a].u4[1] = *(const uint4*)&ldsA[buf][row * 32 + 16 + 8 * half];
        }
        // B fragments: lane (n=l16, khalf=half) holds K in [8*half,+8) and [16+8*half,+8)
        AFrag bfr[4];
#pragma unroll
        for (int b = 0; b < 4; ++b) {
            int col = wn * 64 + b * 16 + l16;
            bfr[b].u4[0] = *(const uint4*)&ldsB[buf][col * 32 + 8 * half];
            bfr[b].u4[1] = *(const uint4*)&ldsB[buf][col * 32 + 16 + 8 * half];
        }

#pragma unroll
        for (int a = 0; a < 2; ++a)
#pragma unroll
            for (int b = 0; b < 4; ++b)
                acc[a][b] = __builtin_amdgcn_wmma_f32_16x16x32_bf16(
                    false, afr[a].v, false, bfr[b].v, (short)0, acc[a][b],
                    false, false);
        buf ^= 1;
    }

    // epilogue: sq = x2 + c2 - 2*dot ; num = 1/(1+sqrt(max(sq,0)))  [ALPHA=1]
    float lsum = 0.f;
#pragma unroll
    for (int a = 0; a < 2; ++a) {
#pragma unroll
        for (int b = 0; b < 4; ++b) {
            int ng = n0 + wn * 64 + b * 16 + l16;
            float cc = c2[ng];
#pragma unroll
            for (int r = 0; r < 8; ++r) {
                int mg = m0 + wm * 32 + a * 16 + r + 8 * half;
                float sq = x2[mg] + cc - 2.0f * acc[a][b][r];
                sq = sq > 0.f ? sq : 0.f;
                float num = 1.0f / (1.0f + sqrtf(sq));
                out[(size_t)mg * NK + ng] = num;
                lsum += num;
            }
        }
    }
#pragma unroll
    for (int off = 16; off > 0; off >>= 1) lsum += __shfl_down(lsum, off, 32);
    if (lane == 0) atomicAdd(gsum, lsum);
}

// ---------------- global normalization ----------------
__global__ void scale_kernel(float* __restrict__ out,
                             const float* __restrict__ gsum) {
    float inv = 1.0f / *gsum;
    const size_t n4 = (size_t)NB * NK / 4;
    float4* o4 = (float4*)out;
    size_t stride = (size_t)gridDim.x * blockDim.x;
    for (size_t j = (size_t)blockIdx.x * blockDim.x + threadIdx.x; j < n4; j += stride) {
        float4 v = o4[j];
        v.x *= inv; v.y *= inv; v.z *= inv; v.w *= inv;
        o4[j] = v;
    }
}

extern "C" void kernel_launch(void* const* d_in, const int* in_sizes, int n_in,
                              void* d_out, int out_size, void* d_ws, size_t ws_size,
                              hipStream_t stream) {
    const float* x = (const float*)d_in[0];
    const float* c = (const float*)d_in[1];
    float* out = (float*)d_out;
    float* ws  = (float*)d_ws;

    float* x2   = ws;             // NB floats
    float* c2   = ws + NB;        // NK floats
    float* gsum = ws + NB + NK;   // 1 float

    zero_kernel<<<1, 1, 0, stream>>>(gsum);
    rownorm_kernel<<<NB / 8, 256, 0, stream>>>(x, x2, NB);
    rownorm_kernel<<<NK / 8, 256, 0, stream>>>(c, c2, NK);

    dim3 grid(NK / 128, NB / 128);  // 8 x 128 = 1024 blocks
    tstudent_gemm_kernel<<<grid, 256, 0, stream>>>(x, c, x2, c2, out, gsum);

    scale_kernel<<<4096, 256, 0, stream>>>(out, gsum);
}
